// MambaNodeMamba_36060545417603
// MI455X (gfx1250) — matrix operand order repaired and checked by reading, not compiled, not run on hardware
//
#include <hip/hip_runtime.h>
#include <hip/hip_bf16.h>

typedef _Float16 f16;
typedef __attribute__((ext_vector_type(16))) _Float16 v16h;
typedef __attribute__((ext_vector_type(8)))  float    v8f;

namespace {
constexpr int kB = 32, kT = 24, kN = 207, kC = 64;
constexpr int kL  = kN * kT;        // 4968
constexpr int kLP = 4976;           // padded to 16*311
constexpr int kNT16 = kLP / 16;     // 311
constexpr long long kMT = (long long)kB * kLP; // 159232 total rows
constexpr int kG  = 24;             // scan chunks per batch
constexpr int kCH = kL / kG;        // 207 steps per chunk (exact)
}

// ---- WMMA fragment loaders (contiguous k = 16*(lane/16) + s in A and B, wave32) -----------

// A fragment: 16x32 f16 tile, row-major source with leading dim `ld`,
// `base` points at (row0, k0).
__device__ __forceinline__ v16h load_a16(const f16* base, int ld, int lane) {
  v16h a;
  const int row = lane & 15;
  const int hi  = (lane >> 4) & 1;
  const int kb1 = hi ? 16 : 0;   // elements 0..7
  const int kb2 = hi ? 24 : 8;   // elements 8..15
  const f16* p = base + (size_t)row * ld;
#pragma unroll
  for (int i = 0; i < 8; ++i) a[i] = p[kb1 + i];
#pragma unroll
  for (int i = 0; i < 8; ++i) a[8 + i] = p[kb2 + i];
  return a;
}

// B fragment for out = act @ W^T : logical B is W^T (32x16 K-major tile).
// W stored row-major (Nout x K), so lane's column reads 16 contiguous K.
__device__ __forceinline__ v16h load_bT16(const f16* W, int ldw, int n0, int k0, int lane) {
  v16h b;
  const int col = n0 + (lane & 15);
  const int kb  = k0 + (((lane >> 4) & 1) ? 16 : 0);
  const f16* p = W + (size_t)col * ldw + kb;
#pragma unroll
  for (int i = 0; i < 16; ++i) b[i] = p[i];
  return b;
}

// ---- kernel 0: convert weights to f16, fold W_dt into W_x ------------------
__global__ void k_prep(const float* __restrict__ W_lin, const float* __restrict__ W_in,
                       const float* __restrict__ W_out, const float* __restrict__ W_x,
                       const float* __restrict__ W_dt,
                       f16* __restrict__ wlin, f16* __restrict__ win,
                       f16* __restrict__ wout, f16* __restrict__ wx3) {
  const int tid = threadIdx.x;
  for (int i = tid; i < 64 * 128; i += 256) wlin[i] = (f16)W_lin[i];
  for (int i = tid; i < 256 * 64; i += 256) win[i]  = (f16)W_in[i];
  for (int i = tid; i < 64 * 128; i += 256) wout[i] = (f16)W_out[i];
  // rows 0..127: Wc[d][j] = sum_r W_dt[d][r] * W_x[r][j]   (dt head folded)
  for (int i = tid; i < 128 * 128; i += 256) {
    const int d = i >> 7, j = i & 127;
    float s = 0.f;
#pragma unroll
    for (int r = 0; r < 4; ++r) s += W_dt[d * 4 + r] * W_x[r * 128 + j];
    wx3[i] = (f16)s;
  }
  // rows 128..159: W_x rows 4..35 (B then C heads)
  for (int i = tid; i < 32 * 128; i += 256) wx3[128 * 128 + i] = (f16)W_x[4 * 128 + i];
}

// ---- kernel 1: gather xcat tile -> GEMM1(+bias) -> GEMM2 -> xm,z -----------
__global__ __launch_bounds__(32) void k_front(
    const float* __restrict__ x, const float* __restrict__ qk,
    const float* __restrict__ b_lin, const f16* __restrict__ wlin,
    const f16* __restrict__ win, f16* __restrict__ xm, f16* __restrict__ zg) {
  __shared__ f16 At[16][128];  // xcat tile (f16)
  __shared__ f16 Sq[16][64];   // seq tile after GEMM1
  const int lane = threadIdx.x;
  const int b  = blockIdx.x / kNT16;
  const int r0 = (blockIdx.x % kNT16) * 16;

  for (int idx = lane; idx < 16 * 128; idx += 32) {
    const int rr = idx >> 7, cc = idx & 127;
    const int l = r0 + rr;
    float v = 0.f;
    if (l < kL) {
      const int n = l / kT, t = l - n * kT;
      const size_t base = ((size_t)(b * kT + t) * kN + n) * kC;
      v = (cc < kC) ? x[base + cc] : qk[base + cc - kC];
    }
    At[rr][cc] = (f16)v;
  }
  __syncthreads();

  v16h a4[4];
#pragma unroll
  for (int kt = 0; kt < 4; ++kt) a4[kt] = load_a16(&At[0][0] + kt * 32, 128, lane);
  const int rhalf = (lane & 16) ? 8 : 0;
  const int ccol  = lane & 15;

  // GEMM1: (16x128) @ wlin^T(128x64) + b_lin  -> Sq
#pragma unroll
  for (int nt = 0; nt < 4; ++nt) {
    v8f acc;
    const float bias = b_lin[nt * 16 + ccol];
#pragma unroll
    for (int vv = 0; vv < 8; ++vv) acc[vv] = bias;
#pragma unroll
    for (int kt = 0; kt < 4; ++kt) {
      v16h bf = load_bT16(wlin, 128, nt * 16, kt * 32, lane);
      acc = __builtin_amdgcn_wmma_f32_16x16x32_f16(false, a4[kt], false, bf,
                                                   (short)0, acc, false, false);
    }
#pragma unroll
    for (int vv = 0; vv < 8; ++vv) Sq[vv + rhalf][nt * 16 + ccol] = (f16)acc[vv];
  }
  __syncthreads();

  // GEMM2: (16x64) @ win^T(64x256) -> xm | z
  v16h a2[2];
#pragma unroll
  for (int kt = 0; kt < 2; ++kt) a2[kt] = load_a16(&Sq[0][0] + kt * 32, 64, lane);
  const size_t rowbase = (size_t)b * kLP + r0;
#pragma unroll
  for (int nt = 0; nt < 16; ++nt) {
    v8f acc;
#pragma unroll
    for (int vv = 0; vv < 8; ++vv) acc[vv] = 0.f;
#pragma unroll
    for (int kt = 0; kt < 2; ++kt) {
      v16h bf = load_bT16(win, 64, nt * 16, kt * 32, lane);
      acc = __builtin_amdgcn_wmma_f32_16x16x32_f16(false, a2[kt], false, bf,
                                                   (short)0, acc, false, false);
    }
    const int col = nt * 16 + ccol;
#pragma unroll
    for (int vv = 0; vv < 8; ++vv) {
      const size_t row = rowbase + vv + rhalf;
      const f16 hv = (f16)acc[vv];
      if (col < 128) xm[row * 128 + col] = hv;
      else           zg[row * 128 + (col - 128)] = hv;
    }
  }
}

// ---- kernel 2: depthwise causal conv (k=4) + SiLU -> xc (pad rows -> 0) ----
__global__ void k_conv(const f16* __restrict__ xm, const float* __restrict__ conv_w,
                       const float* __restrict__ conv_b, f16* __restrict__ xc) {
  const size_t idx = (size_t)blockIdx.x * blockDim.x + threadIdx.x;
  if (idx >= (size_t)kB * kLP * 128) return;
  const int d = (int)(idx & 127);
  const size_t rl = idx >> 7;            // b*kLP + l
  const int l = (int)(rl % kLP);
  float out = 0.f;
  if (l < kL) {
    float acc = conv_b[d];
#pragma unroll
    for (int k = 0; k < 4; ++k) {
      const int lm = l - 3 + k;
      if (lm >= 0) acc = fmaf((float)xm[(rl - (size_t)l + lm) * 128 + d], conv_w[d * 4 + k], acc);
    }
    out = acc / (1.f + __expf(-acc));    // SiLU
  }
  xc[idx] = (f16)out;
}

// ---- kernel 3: GEMM3 xc @ wx3^T(128x160); softplus dt head -> dtbc ---------
__global__ __launch_bounds__(32) void k_gemm3(const f16* __restrict__ xc,
                                              const f16* __restrict__ wx3,
                                              const float* __restrict__ b_dt,
                                              f16* __restrict__ dtbc) {
  const int lane = threadIdx.x;
  const size_t r0 = (size_t)blockIdx.x * 16;
  v16h a4[4];
#pragma unroll
  for (int kt = 0; kt < 4; ++kt) a4[kt] = load_a16(xc + r0 * 128 + kt * 32, 128, lane);
  const int rhalf = (lane & 16) ? 8 : 0;
  const int ccol  = lane & 15;
#pragma unroll
  for (int nt = 0; nt < 10; ++nt) {
    v8f acc;
#pragma unroll
    for (int vv = 0; vv < 8; ++vv) acc[vv] = 0.f;
#pragma unroll
    for (int kt = 0; kt < 4; ++kt) {
      v16h bf = load_bT16(wx3, 128, nt * 16, kt * 32, lane);
      acc = __builtin_amdgcn_wmma_f32_16x16x32_f16(false, a4[kt], false, bf,
                                                   (short)0, acc, false, false);
    }
    const int col = nt * 16 + ccol;
#pragma unroll
    for (int vv = 0; vv < 8; ++vv) {
      const size_t row = r0 + vv + rhalf;
      float val = acc[vv];
      if (col < 128) {                    // dt head: + b_dt, softplus
        val += b_dt[col];
        val = (val > 20.f) ? val : log1pf(__expf(val));
      }
      dtbc[row * 160 + col] = (f16)val;
    }
  }
}

// ---- scan pass 1: per-chunk (transition product P, end state H) ------------
// Per (d,s): a_l = exp(dt_l*A[d,s])  =>  prod_l a_l = exp(A[d,s] * sum_l dt_l)
__global__ __launch_bounds__(128) void k_scan1(const f16* __restrict__ dtbc,
                                               const f16* __restrict__ xc,
                                               const float* __restrict__ A_log,
                                               float* __restrict__ Pbuf,
                                               float* __restrict__ Hbuf) {
  const int b = blockIdx.x / kG, g = blockIdx.x % kG;
  const int d = threadIdx.x;
  float Ad[16];
#pragma unroll
  for (int s = 0; s < 16; ++s) Ad[s] = -__expf(A_log[d * 16 + s]);
  float h[16];
#pragma unroll
  for (int s = 0; s < 16; ++s) h[s] = 0.f;
  float sdt = 0.f;
  const int l0 = g * kCH, l1 = l0 + kCH;
  for (int l = l0; l < l1; ++l) {
    const size_t row = (size_t)b * kLP + l;
    __builtin_prefetch(dtbc + (row + 4) * 160, 0, 3);
    const float dt = (float)dtbc[row * 160 + d];
    const float xv = (float)xc[row * 128 + d];
    const v16h Bv = *(const v16h*)(dtbc + row * 160 + 128);  // B_t (16 x f16)
    const float dtx = dt * xv;
    sdt += dt;
#pragma unroll
    for (int s = 0; s < 16; ++s)
      h[s] = fmaf(__expf(dt * Ad[s]), h[s], dtx * (float)Bv[s]);
  }
  const int base = ((b * kG + g) * 128 + d) * 16;
#pragma unroll
  for (int s = 0; s < 16; ++s) {
    Pbuf[base + s] = __expf(Ad[s] * sdt);
    Hbuf[base + s] = h[s];
  }
}

// ---- scan pass 2: serial combine over chunks -> initial state per chunk ----
__global__ __launch_bounds__(128) void k_scan2(const float* __restrict__ Pbuf,
                                               const float* __restrict__ Hbuf,
                                               float* __restrict__ H0buf) {
  const int b = blockIdx.x, d = threadIdx.x;
  float h0[16];
#pragma unroll
  for (int s = 0; s < 16; ++s) h0[s] = 0.f;
  for (int g = 0; g < kG; ++g) {
    const int base = ((b * kG + g) * 128 + d) * 16;
#pragma unroll
    for (int s = 0; s < 16; ++s) {
      H0buf[base + s] = h0[s];
      h0[s] = fmaf(Pbuf[base + s], h0[s], Hbuf[base + s]);
    }
  }
}

// ---- scan pass 3: recurrence with true h0, y + D residual + SiLU(z) gate ---
__global__ __launch_bounds__(128) void k_scan3(const f16* __restrict__ dtbc,
                                               const f16* __restrict__ xc,
                                               const f16* __restrict__ zg,
                                               const float* __restrict__ A_log,
                                               const float* __restrict__ D_param,
                                               const float* __restrict__ H0buf,
                                               f16* __restrict__ ycomb) {
  const int b = blockIdx.x / kG, g = blockIdx.x % kG;
  const int d = threadIdx.x;
  float Ad[16];
#pragma unroll
  for (int s = 0; s < 16; ++s) Ad[s] = -__expf(A_log[d * 16 + s]);
  const float Dp = D_param[d];
  float h[16];
  const int base = ((b * kG + g) * 128 + d) * 16;
#pragma unroll
  for (int s = 0; s < 16; ++s) h[s] = H0buf[base + s];
  const int l0 = g * kCH, l1 = l0 + kCH;
  for (int l = l0; l < l1; ++l) {
    const size_t row = (size_t)b * kLP + l;
    __builtin_prefetch(dtbc + (row + 4) * 160, 0, 3);
    const float dt = (float)dtbc[row * 160 + d];
    const float xv = (float)xc[row * 128 + d];
    const v16h Bv = *(const v16h*)(dtbc + row * 160 + 128);  // B_t
    const v16h Cv = *(const v16h*)(dtbc + row * 160 + 144);  // C_t
    const float dtx = dt * xv;
    float y = 0.f;
#pragma unroll
    for (int s = 0; s < 16; ++s) {
      h[s] = fmaf(__expf(dt * Ad[s]), h[s], dtx * (float)Bv[s]);
      y = fmaf(h[s], (float)Cv[s], y);
    }
    y = fmaf(xv, Dp, y);
    const float zv = (float)zg[row * 128 + d];
    y *= zv / (1.f + __expf(-zv));      // * SiLU(z)
    ycomb[row * 128 + d] = (f16)y;
  }
  if (g == kG - 1) {                    // zero pad rows for GEMM4
    for (int l = kL; l < kLP; ++l)
      ycomb[((size_t)b * kLP + l) * 128 + d] = (f16)0.f;
  }
}

// ---- kernel 5: GEMM4 ycomb @ wout^T(128x64) -> ymm (f32) -------------------
__global__ __launch_bounds__(32) void k_gemm4(const f16* __restrict__ ycomb,
                                              const f16* __restrict__ wout,
                                              float* __restrict__ ymm) {
  const int lane = threadIdx.x;
  const size_t r0 = (size_t)blockIdx.x * 16;
  v16h a4[4];
#pragma unroll
  for (int kt = 0; kt < 4; ++kt) a4[kt] = load_a16(ycomb + r0 * 128 + kt * 32, 128, lane);
  const int rhalf = (lane & 16) ? 8 : 0;
  const int ccol  = lane & 15;
#pragma unroll
  for (int nt = 0; nt < 4; ++nt) {
    v8f acc;
#pragma unroll
    for (int vv = 0; vv < 8; ++vv) acc[vv] = 0.f;
#pragma unroll
    for (int kt = 0; kt < 4; ++kt) {
      v16h bf = load_bT16(wout, 128, nt * 16, kt * 32, lane);
      acc = __builtin_amdgcn_wmma_f32_16x16x32_f16(false, a4[kt], false, bf,
                                                   (short)0, acc, false, false);
    }
    const int col = nt * 16 + ccol;
#pragma unroll
    for (int vv = 0; vv < 8; ++vv)
      ymm[(r0 + vv + rhalf) * 64 + col] = acc[vv];
  }
}

// ---- kernel 6: group-norm statistics per (b, group) ------------------------
__global__ __launch_bounds__(256) void k_stats(const float* __restrict__ ymm,
                                               const float* __restrict__ x,
                                               float* __restrict__ stats) {
  const int b = blockIdx.x >> 2, g = blockIdx.x & 3;
  float s = 0.f, s2 = 0.f;
  for (int i = threadIdx.x; i < 16 * kL; i += 256) {
    const int q = i / kL;
    const int l = i - q * kL;
    const int c = g * 16 + q;
    const int t = l % kT, n = l / kT;
    const float hv = ymm[((size_t)b * kLP + l) * 64 + c] +
                     x[((size_t)(b * kT + t) * kN + n) * 64 + c];
    s += hv; s2 += hv * hv;
  }
  __shared__ float rs[256], rq[256];
  rs[threadIdx.x] = s; rq[threadIdx.x] = s2;
  __syncthreads();
  for (int off = 128; off > 0; off >>= 1) {
    if (threadIdx.x < off) {
      rs[threadIdx.x] += rs[threadIdx.x + off];
      rq[threadIdx.x] += rq[threadIdx.x + off];
    }
    __syncthreads();
  }
  if (threadIdx.x == 0) {
    const float inv = 1.f / (16.f * kL);
    const float mu = rs[0] * inv;
    const float var = rq[0] * inv - mu * mu;
    stats[blockIdx.x * 2]     = mu;
    stats[blockIdx.x * 2 + 1] = rsqrtf(var + 1e-5f);
  }
}

// ---- kernel 7: residual + normalize + affine -> out ------------------------
__global__ void k_out(const float* __restrict__ ymm, const float* __restrict__ x,
                      const float* __restrict__ stats, const float* __restrict__ gn_w,
                      const float* __restrict__ gn_b, float* __restrict__ out) {
  const size_t idx = (size_t)blockIdx.x * 256 + threadIdx.x;
  if (idx >= (size_t)kB * kT * kN * kC) return;
  const int c = (int)(idx & 63);
  size_t r = idx >> 6;
  const int n = (int)(r % kN); r /= kN;
  const int t = (int)(r % kT);
  const int b = (int)(r / kT);
  const int l = n * kT + t;
  const int bg = b * 4 + (c >> 4);
  const float mu = stats[bg * 2], rv = stats[bg * 2 + 1];
  const float hv = ymm[((size_t)b * kLP + l) * 64 + c] + x[idx];
  out[idx] = (hv - mu) * rv * gn_w[c] + gn_b[c];
}

extern "C" void kernel_launch(void* const* d_in, const int* in_sizes, int n_in,
                              void* d_out, int out_size, void* d_ws, size_t ws_size,
                              hipStream_t stream) {
  (void)in_sizes; (void)n_in; (void)out_size;
  const float* x      = (const float*)d_in[0];
  const float* qk     = (const float*)d_in[1];
  const float* W_lin  = (const float*)d_in[2];
  const float* b_lin  = (const float*)d_in[3];
  const float* W_in   = (const float*)d_in[4];
  const float* conv_w = (const float*)d_in[5];
  const float* conv_b = (const float*)d_in[6];
  const float* W_x    = (const float*)d_in[7];
  const float* W_dt   = (const float*)d_in[8];
  const float* b_dt   = (const float*)d_in[9];
  const float* A_log  = (const float*)d_in[10];
  const float* D_par  = (const float*)d_in[11];
  const float* W_out  = (const float*)d_in[12];
  const float* gn_w   = (const float*)d_in[13];
  const float* gn_b   = (const float*)d_in[14];
  float* out = (float*)d_out;

  char* ws = (char*)d_ws;
  const size_t SZ128h = (size_t)kMT * 128 * sizeof(f16);  // 40.76 MB
  const size_t SZ160h = (size_t)kMT * 160 * sizeof(f16);  // 50.95 MB
  const size_t SZscan = (size_t)kB * kG * 128 * 16 * sizeof(float); // 6.29 MB
  f16*   xm    = (f16*)(ws);                      // dead after k_conv
  f16*   zg    = (f16*)(ws + SZ128h);
  f16*   xc    = (f16*)(ws + 2 * SZ128h);
  f16*   dtbc  = (f16*)(ws + 3 * SZ128h);
  f16*   ycomb = (f16*)(ws + 3 * SZ128h + SZ160h);
  float* ymm   = (float*)(ws);                    // aliases xm (same byte size)
  size_t woff = 4 * SZ128h + SZ160h;
  float* Pbuf  = (float*)(ws + woff); woff += SZscan;
  float* Hbuf  = (float*)(ws + woff); woff += SZscan;
  float* H0buf = (float*)(ws + woff); woff += SZscan;
  f16* wlin = (f16*)(ws + woff); woff += 64 * 128 * sizeof(f16);
  f16* win  = (f16*)(ws + woff); woff += 256 * 64 * sizeof(f16);
  f16* wout = (f16*)(ws + woff); woff += 64 * 128 * sizeof(f16);
  f16* wx3  = (f16*)(ws + woff); woff += 160 * 128 * sizeof(f16);
  float* stats = (float*)(ws + woff); woff += (size_t)kB * 4 * 2 * sizeof(float);
  if (ws_size < woff) return;  // workspace too small (deterministic no-op)

  const int gemmBlocks = (int)(kMT / 16);  // 9952
  k_prep<<<1, 256, 0, stream>>>(W_lin, W_in, W_out, W_x, W_dt, wlin, win, wout, wx3);
  k_front<<<kB * kNT16, 32, 0, stream>>>(x, qk, b_lin, wlin, win, xm, zg);
  k_conv<<<(int)(((size_t)kB * kLP * 128 + 255) / 256), 256, 0, stream>>>(xm, conv_w, conv_b, xc);
  k_gemm3<<<gemmBlocks, 32, 0, stream>>>(xc, wx3, b_dt, dtbc);
  k_scan1<<<kB * kG, 128, 0, stream>>>(dtbc, xc, A_log, Pbuf, Hbuf);
  k_scan2<<<kB, 128, 0, stream>>>(Pbuf, Hbuf, H0buf);
  k_scan3<<<kB * kG, 128, 0, stream>>>(dtbc, xc, zg, A_log, D_par, H0buf, ycomb);
  k_gemm4<<<gemmBlocks, 32, 0, stream>>>(ycomb, wout, ymm);
  k_stats<<<kB * 4, 256, 0, stream>>>(ymm, x, stats);
  k_out<<<(int)(((size_t)kB * kT * kN * kC + 255) / 256), 256, 0, stream>>>(
      ymm, x, stats, gn_w, gn_b, out);
}
